// PatchGCN_85796266704896
// MI455X (gfx1250) — matrix operand order repaired and checked by reading, not compiled, not run on hardware
//
#include <hip/hip_runtime.h>
#include <math.h>

#define NNODES 100000
#define NEDGES 600000
#define CIN    1024
#define HD     128
#define NCLS   4
#define NTYP   7
#define LDIM   512   // 4*HD

typedef __attribute__((ext_vector_type(16))) __bf16 v16bf;
typedef __attribute__((ext_vector_type(8)))  float  v8f;

enum { ACT_NONE = 0, ACT_RELU = 1, ACT_TANH = 2, ACT_SIGMUL = 3 };

__device__ __forceinline__ unsigned short f2bf(float f) {
  unsigned u = __float_as_uint(f);
  u += 0x7FFFu + ((u >> 16) & 1u);           // round-to-nearest-even
  return (unsigned short)(u >> 16);
}
__device__ __forceinline__ unsigned pkbf(float lo, float hi) {
  return (unsigned)f2bf(lo) | ((unsigned)f2bf(hi) << 16);
}
__device__ __forceinline__ unsigned fkey(float f) {  // order-preserving float->uint
  unsigned u = __float_as_uint(f);
  return (u & 0x80000000u) ? ~u : (u | 0x80000000u);
}
__device__ __forceinline__ float fdekey(unsigned k) {
  return (k & 0x80000000u) ? __uint_as_float(k ^ 0x80000000u) : __uint_as_float(~k);
}
__device__ __forceinline__ float wsum(float v) {
#pragma unroll
  for (int o = 16; o > 0; o >>= 1) v += __shfl_xor(v, o, 32);
  return v;
}
__device__ __forceinline__ float wmaxr(float v) {
#pragma unroll
  for (int o = 16; o > 0; o >>= 1) v = fmaxf(v, __shfl_xor(v, o, 32));
  return v;
}

// ---------------------------------------------------------------------------
// Generic bf16 WMMA GEMM:  C[M,Nc] = act(A[M,K] * B[K,Nc] + bias)
// Block: 256 threads (8 waves). Tile BM=128, BN=64, BK=32.
// Grid: (Nc/BN, M/BM) -- column tiles on x so blocks sharing an A row-slab
// are dispatched adjacently and hit L2 for A.
// Double-buffered LDS staging; f32->bf16 conversion done in registers,
// stored packed (2 bf16 / u32) so stages are ds_store_b128 pairs and
// fragment reads are contiguous ds_load_b128 pairs (ISA 7.12.2 layouts).
// ---------------------------------------------------------------------------
#define BM 128
#define BN 64
#define BK 32

__global__ __launch_bounds__(256)
void gemm_wmma_bf16(const float* __restrict__ A, int lda,
                    const float* __restrict__ Bw, int ldb,
                    const float* __restrict__ bias,
                    float* __restrict__ C, int ldc,
                    int M, int K, int act)
{
  __shared__ unsigned sA[2][(BM / 16) * 32 * 8];  // 2 x 2048 u32 = 16 KB
  __shared__ unsigned sB[2][(BN / 16) * 32 * 8];  // 2 x 1024 u32 = 8 KB

  const int tid  = threadIdx.x;
  const int lane = tid & 31;
  const int wave = tid >> 5;
  const int colBase = blockIdx.x * BN;   // column tiles vary fastest
  const int rowBase = blockIdx.y * BM;

  const int wm = wave >> 1;   // 0..3  (32-row slab)
  const int wn = wave & 1;    // 0..1  (32-col slab)

  v8f acc[2][2] = {};

  // A staging slot: thread t fills lane-slot (subtile sm = wave, lane)
  const int  a_row = rowBase + wave * 16 + (lane & 15);
  const int  a_kb  = (lane >> 4) * 8;      // 0 or 8
  const bool a_ok  = a_row < M;
  // B staging slot (threads < 128)
  const int  b_col = colBase + (wave & 3) * 16 + (lane & 15);
  const int  b_kb  = (lane >> 4) * 8;      // 0 or 8
  const bool b_th  = tid < 128;

  unsigned ra[8] = {};
  unsigned rb[8] = {};

  auto loadA = [&](int k0) {
    if (a_ok) {
      const float* ap = A + (size_t)a_row * lda + k0 + a_kb;
      float4 r0 = *(const float4*)(ap);
      float4 r1 = *(const float4*)(ap + 4);
      float4 r2 = *(const float4*)(ap + 16);
      float4 r3 = *(const float4*)(ap + 20);
      ra[0] = pkbf(r0.x, r0.y); ra[1] = pkbf(r0.z, r0.w);
      ra[2] = pkbf(r1.x, r1.y); ra[3] = pkbf(r1.z, r1.w);
      ra[4] = pkbf(r2.x, r2.y); ra[5] = pkbf(r2.z, r2.w);
      ra[6] = pkbf(r3.x, r3.y); ra[7] = pkbf(r3.z, r3.w);
    } else {
#pragma unroll
      for (int e = 0; e < 8; e++) ra[e] = 0u;
    }
  };
  auto loadB = [&](int k0) {
    if (b_th) {
      const float* bp = Bw + (size_t)(k0 + b_kb) * ldb + b_col;
#pragma unroll
      for (int e = 0; e < 8; e++)
        rb[e] = pkbf(bp[(size_t)(2 * e + (e & 4) * 2) * ldb], bp[(size_t)(2 * e + (e & 4) * 2 + 1) * ldb]);
    }
  };
  auto stage = [&](int bsel) {
    uint4* da = (uint4*)&sA[bsel][tid * 8];
    da[0] = make_uint4(ra[0], ra[1], ra[2], ra[3]);
    da[1] = make_uint4(ra[4], ra[5], ra[6], ra[7]);
    if (b_th) {
      uint4* db = (uint4*)&sB[bsel][tid * 8];
      db[0] = make_uint4(rb[0], rb[1], rb[2], rb[3]);
      db[1] = make_uint4(rb[4], rb[5], rb[6], rb[7]);
    }
  };

  loadA(0); loadB(0); stage(0);
  __syncthreads();

  int buf = 0;
  for (int k0 = 0; k0 < K; k0 += BK) {
    const bool havenext = (k0 + BK) < K;
    if (havenext) { loadA(k0 + BK); loadB(k0 + BK); }  // global loads in flight

    v16bf af[2], bf[2];
#pragma unroll
    for (int i = 0; i < 2; i++)
      af[i] = *reinterpret_cast<const v16bf*>(&sA[buf][((wm * 2 + i) * 32 + lane) * 8]);
#pragma unroll
    for (int j = 0; j < 2; j++)
      bf[j] = *reinterpret_cast<const v16bf*>(&sB[buf][((wn * 2 + j) * 32 + lane) * 8]);

#pragma unroll
    for (int i = 0; i < 2; i++)
#pragma unroll
      for (int j = 0; j < 2; j++)
        acc[i][j] = __builtin_amdgcn_wmma_f32_16x16x32_bf16(
            false, af[i], false, bf[j], (short)0, acc[i][j], false, false);

    if (havenext) stage(buf ^ 1);
    __syncthreads();
    buf ^= 1;
  }

  // ---- epilogue: C/D layout = lane (N=lane&15), VGPR r -> M = r + (lane>=16)*8
#pragma unroll
  for (int i = 0; i < 2; i++) {
#pragma unroll
    for (int j = 0; j < 2; j++) {
      const int col = colBase + (wn * 2 + j) * 16 + (lane & 15);
      const float bv = bias ? bias[col] : 0.f;
#pragma unroll
      for (int r = 0; r < 8; r++) {
        const int row = rowBase + (wm * 2 + i) * 16 + ((lane >> 4) * 8) + r;
        if (row < M) {
          float v = acc[i][j][r] + bv;
          const size_t o = (size_t)row * ldc + col;
          if (act == ACT_RELU)        v = v > 0.f ? v : 0.f;
          else if (act == ACT_TANH)   v = tanhf(v);
          else if (act == ACT_SIGMUL) v = (1.f / (1.f + __expf(-v))) * C[o];
          C[o] = v;
        }
      }
    }
  }
}

// ---------------------------------------------------------------------------
// GENConv scatter-softmax aggregation
// ---------------------------------------------------------------------------
__global__ __launch_bounds__(256)
void k_zero3(float* agg, unsigned* mxu, float* den, int n) {
  int i = blockIdx.x * 256 + threadIdx.x;
  if (i < n) { agg[i] = 0.f; den[i] = 0.f; mxu[i] = 0u; }
}

__global__ __launch_bounds__(256)
void k_smax(const float* __restrict__ X, int ldx,
            const int* __restrict__ src, const int* __restrict__ dst,
            unsigned* __restrict__ mxu, const float* __restrict__ tptr, long total) {
  long idx = (long)blockIdx.x * 256 + threadIdx.x;
  if (idx >= total) return;
  int e = (int)(idx >> 7), h = (int)(idx & 127);
  float t = tptr[0];
  float m = X[(size_t)src[e] * ldx + h];
  m = (m > 0.f ? m : 0.f) + 1e-7f;
  atomicMax(&mxu[(size_t)dst[e] * HD + h], fkey(m * t));
}

__global__ __launch_bounds__(256)
void k_ssum(const float* __restrict__ X, int ldx,
            const int* __restrict__ src, const int* __restrict__ dst,
            const unsigned* __restrict__ mxu,
            float* __restrict__ den, float* __restrict__ agg,
            const float* __restrict__ tptr, long total) {
  long idx = (long)blockIdx.x * 256 + threadIdx.x;
  if (idx >= total) return;
  int e = (int)(idx >> 7), h = (int)(idx & 127);
  float t = tptr[0];
  float m = X[(size_t)src[e] * ldx + h];
  m = (m > 0.f ? m : 0.f) + 1e-7f;
  size_t o = (size_t)dst[e] * HD + h;
  float ex = __expf(m * t - fdekey(mxu[o]));
  atomicAdd(&den[o], ex);
  atomicAdd(&agg[o], m * ex);
}

__global__ __launch_bounds__(256)
void k_finish_agg(float* __restrict__ agg, const float* __restrict__ den,
                  const float* __restrict__ X, int ldx, int n) {
  int i = blockIdx.x * 256 + threadIdx.x;
  if (i >= n) return;
  int node = i >> 7, h = i & 127;
  agg[i] = agg[i] / (den[i] + 1e-16f) + X[(size_t)node * ldx + h];
}

// ---------------------------------------------------------------------------
// LayerNorm (+ReLU) kernels: one wave per row
// ---------------------------------------------------------------------------
__global__ __launch_bounds__(256)
void k_ln_relu256(float* __restrict__ Hm, const float* __restrict__ g,
                  const float* __restrict__ b, int n) {
  int lane = threadIdx.x & 31, wave = threadIdx.x >> 5;
  int row = blockIdx.x * 8 + wave;
  if (row >= n) return;
  float* p = Hm + (size_t)row * 256;
  float v[8], s = 0.f, s2 = 0.f;
#pragma unroll
  for (int j = 0; j < 8; j++) { v[j] = p[lane + 32 * j]; s += v[j]; s2 += v[j] * v[j]; }
  s = wsum(s); s2 = wsum(s2);
  float mu = s * (1.f / 256.f);
  float var = s2 * (1.f / 256.f) - mu * mu;
  float inv = rsqrtf(var + 1e-5f);
#pragma unroll
  for (int j = 0; j < 8; j++) {
    int c = lane + 32 * j;
    float y = (v[j] - mu) * inv * g[c] + b[c];
    p[c] = y > 0.f ? y : 0.f;
  }
}

__global__ __launch_bounds__(256)
void k_resid128(const float* __restrict__ conv, const float* __restrict__ g,
                const float* __restrict__ b, const float* __restrict__ prev,
                float* __restrict__ out, int n) {
  int lane = threadIdx.x & 31, wave = threadIdx.x >> 5;
  int row = blockIdx.x * 8 + wave;
  if (row >= n) return;
  const float* p = conv + (size_t)row * HD;
  float v[4], s = 0.f, s2 = 0.f;
#pragma unroll
  for (int j = 0; j < 4; j++) { v[j] = p[lane + 32 * j]; s += v[j]; s2 += v[j] * v[j]; }
  s = wsum(s); s2 = wsum(s2);
  float mu = s * (1.f / 128.f);
  float var = s2 * (1.f / 128.f) - mu * mu;
  float inv = rsqrtf(var + 1e-5f);
#pragma unroll
  for (int j = 0; j < 4; j++) {
    int c = lane + 32 * j;
    float y = (v[j] - mu) * inv * g[c] + b[c];
    y = y > 0.f ? y : 0.f;
    out[(size_t)row * LDIM + c] = y + prev[(size_t)row * LDIM + c];
  }
}

// ---------------------------------------------------------------------------
// Attention head
// ---------------------------------------------------------------------------
__global__ __launch_bounds__(256)
void k_attnA(const float* __restrict__ AG, const float* __restrict__ wc,
             const float* __restrict__ bc, float* __restrict__ araw, int n) {
  int lane = threadIdx.x & 31, wave = threadIdx.x >> 5;
  int node = blockIdx.x * 8 + wave;
  if (node >= n) return;
  const float* p = AG + (size_t)node * LDIM;
  float a0 = 0.f, a1 = 0.f, a2 = 0.f, a3 = 0.f;
#pragma unroll
  for (int j = 0; j < 16; j++) {
    int r = lane + 32 * j;
    float v = p[r];
    const float* w = wc + (size_t)r * 4;
    a0 += v * w[0]; a1 += v * w[1]; a2 += v * w[2]; a3 += v * w[3];
  }
  a0 = wsum(a0); a1 = wsum(a1); a2 = wsum(a2); a3 = wsum(a3);
  if (lane == 0) {
    float* o = araw + (size_t)node * 4;
    o[0] = a0 + bc[0]; o[1] = a1 + bc[1]; o[2] = a2 + bc[2]; o[3] = a3 + bc[3];
  }
}

__global__ __launch_bounds__(256)
void k_zero_stats(unsigned* cmaxu, float* csum, float* Mbuf) {
  int t = threadIdx.x;
  if (t < 4) { cmaxu[t] = 0u; csum[t] = 0.f; }
  for (int i = t; i < NCLS * LDIM; i += 256) Mbuf[i] = 0.f;
}

__global__ __launch_bounds__(256)
void k_cmax(const float* __restrict__ araw, unsigned* __restrict__ cmaxu, int n) {
  int lane = threadIdx.x & 31, wave = threadIdx.x >> 5;
  int node = blockIdx.x * 256 + threadIdx.x;
  __shared__ float sm[4][8];
#pragma unroll
  for (int c = 0; c < 4; c++) {
    float v = (node < n) ? araw[(size_t)node * 4 + c] : -3.4e38f;
    v = wmaxr(v);
    if (lane == 0) sm[c][wave] = v;
  }
  __syncthreads();
  if (threadIdx.x < 4) {
    float m = sm[threadIdx.x][0];
#pragma unroll
    for (int w = 1; w < 8; w++) m = fmaxf(m, sm[threadIdx.x][w]);
    atomicMax(&cmaxu[threadIdx.x], fkey(m));
  }
}

__global__ __launch_bounds__(256)
void k_csum(const float* __restrict__ araw, const unsigned* __restrict__ cmaxu,
            float* __restrict__ csum, int n) {
  int lane = threadIdx.x & 31, wave = threadIdx.x >> 5;
  int node = blockIdx.x * 256 + threadIdx.x;
  __shared__ float sm[4][8];
#pragma unroll
  for (int c = 0; c < 4; c++) {
    float v = (node < n) ? __expf(araw[(size_t)node * 4 + c] - fdekey(cmaxu[c])) : 0.f;
    v = wsum(v);
    if (lane == 0) sm[c][wave] = v;
  }
  __syncthreads();
  if (threadIdx.x < 4) {
    float s = 0.f;
#pragma unroll
    for (int w = 0; w < 8; w++) s += sm[threadIdx.x][w];
    atomicAdd(&csum[threadIdx.x], s);
  }
}

__global__ __launch_bounds__(256)
void k_wnorm(float* __restrict__ araw, const unsigned* __restrict__ cmaxu,
             const float* __restrict__ csum, int n4) {
  int i = blockIdx.x * 256 + threadIdx.x;
  if (i >= n4) return;
  int c = i & 3;
  araw[i] = __expf(araw[i] - fdekey(cmaxu[c])) / csum[c];
}

__global__ __launch_bounds__(256)
void k_pool(const float* __restrict__ hpath, const float* __restrict__ w,
            float* __restrict__ Mbuf, int n) {
  __shared__ float accM[NCLS * LDIM];  // 8 KB
  int t = threadIdx.x;
  for (int i = t; i < NCLS * LDIM; i += 256) accM[i] = 0.f;
  __syncthreads();
  int nodeBase = blockIdx.x * 256;
  int h0 = t, h1 = t + 256;
  for (int k = 0; k < 256; k++) {
    int node = nodeBase + k;
    if (node >= n) break;
    float hp0 = hpath[(size_t)node * LDIM + h0];
    float hp1 = hpath[(size_t)node * LDIM + h1];
    const float* ww = w + (size_t)node * 4;
    float w0 = ww[0], w1 = ww[1], w2 = ww[2], w3 = ww[3];
    accM[0 * LDIM + h0] += w0 * hp0; accM[0 * LDIM + h1] += w0 * hp1;
    accM[1 * LDIM + h0] += w1 * hp0; accM[1 * LDIM + h1] += w1 * hp1;
    accM[2 * LDIM + h0] += w2 * hp0; accM[2 * LDIM + h1] += w2 * hp1;
    accM[3 * LDIM + h0] += w3 * hp0; accM[3 * LDIM + h1] += w3 * hp1;
  }
  __syncthreads();
  for (int i = t; i < NCLS * LDIM; i += 256) atomicAdd(&Mbuf[i], accM[i]);
}

__global__ void k_tf(const float* __restrict__ typ, const float* __restrict__ tw,
                     const float* __restrict__ tb, float* __restrict__ tf) {
  int h = threadIdx.x;  // 128 threads
  float s = tb[h];
#pragma unroll
  for (int j = 0; j < NTYP; j++) s += typ[j] * tw[j * HD + h];
  tf[h] = s;
}

__global__ void k_final(const float* __restrict__ Mbuf, const float* __restrict__ tf,
                        const float* __restrict__ clsw, const float* __restrict__ clsb,
                        float* __restrict__ out) {
  int lane = threadIdx.x & 31, c = threadIdx.x >> 5;  // 128 threads = 4 waves
  float s = 0.f;
  for (int j = lane; j < LDIM + HD; j += 32) {
    float mv = (j < LDIM) ? Mbuf[c * LDIM + j] : tf[j - LDIM];
    s += mv * clsw[(size_t)c * (LDIM + HD) + j];
  }
  s = wsum(s);
  __shared__ float lg[4];
  if (lane == 0) lg[c] = s + clsb[c];
  __syncthreads();
  if (threadIdx.x == 0) {
    float mx = lg[0]; int am = 0;
    for (int i = 1; i < 4; i++) if (lg[i] > mx) { mx = lg[i]; am = i; }
    float e[4], den = 0.f;
    for (int i = 0; i < 4; i++) { e[i] = __expf(lg[i] - mx); den += e[i]; }
    for (int i = 0; i < 4; i++) out[i] = lg[i];
    for (int i = 0; i < 4; i++) out[4 + i] = e[i] / den;
    out[8] = (float)am;
  }
}

// ---------------------------------------------------------------------------
extern "C" void kernel_launch(void* const* d_in, const int* in_sizes, int n_in,
                              void* d_out, int out_size, void* d_ws, size_t ws_size,
                              hipStream_t stream) {
  (void)in_sizes; (void)n_in; (void)out_size; (void)ws_size;
  const float* x    = (const float*)d_in[0];
  const float* typ  = (const float*)d_in[1];
  const float* fcw  = (const float*)d_in[2];
  const float* fcb  = (const float*)d_in[3];
  const float* cw1  = (const float*)d_in[4];
  const float* cb1  = (const float*)d_in[5];
  const float* cg1  = (const float*)d_in[6];
  const float* cbe1 = (const float*)d_in[7];
  const float* cw2  = (const float*)d_in[8];
  const float* cb2  = (const float*)d_in[9];
  const float* ct   = (const float*)d_in[10];
  const float* dgg  = (const float*)d_in[11];
  const float* dgb  = (const float*)d_in[12];
  const float* wa   = (const float*)d_in[13];
  const float* ba   = (const float*)d_in[14];
  const float* wb   = (const float*)d_in[15];
  const float* bb   = (const float*)d_in[16];
  const float* wc   = (const float*)d_in[17];
  const float* bc   = (const float*)d_in[18];
  const float* tw   = (const float*)d_in[19];
  const float* tb   = (const float*)d_in[20];
  const float* clw  = (const float*)d_in[21];
  const float* clb  = (const float*)d_in[22];
  const int*   ei   = (const int*)d_in[23];
  const int* src = ei;
  const int* dst = ei + NEDGES;
  float* out = (float*)d_out;

  float* W = (float*)d_ws;
  // workspace layout (in floats)
  const size_t oHP   = 0;                              // N*512  h_path
  const size_t oAGG  = (size_t)NNODES * LDIM;          // N*128
  const size_t oMXU  = oAGG  + (size_t)NNODES * HD;    // N*128 (uint keys)
  const size_t oDEN  = oMXU  + (size_t)NNODES * HD;    // N*128
  const size_t oHMID = oDEN  + (size_t)NNODES * HD;    // N*256
  const size_t oCONV = oHMID + (size_t)NNODES * 2 * HD;// N*128
  const size_t oAG   = oAGG;                           // alias: N*512 fits in conv scratch
  const size_t oARAW = oAGG  + (size_t)NNODES * (LDIM + HD);  // N*4
  const size_t oCMAX = oARAW + (size_t)NNODES * 4;     // 4 (uint)
  const size_t oCSUM = oCMAX + 4;                      // 4
  const size_t oMB   = oCSUM + 4;                      // 4*512
  const size_t oTF   = oMB   + (size_t)NCLS * LDIM;    // 128

  float*    HP   = W + oHP;
  float*    AGG  = W + oAGG;
  unsigned* MXU  = (unsigned*)(W + oMXU);
  float*    DEN  = W + oDEN;
  float*    HMID = W + oHMID;
  float*    CONV = W + oCONV;
  float*    AG   = W + oAG;
  float*    ARAW = W + oARAW;
  unsigned* CMAX = (unsigned*)(W + oCMAX);
  float*    CSUM = W + oCSUM;
  float*    MB   = W + oMB;
  float*    TF   = W + oTF;

  const int GM = (NNODES + BM - 1) / BM;               // 782 row-tiles
  const long EH = (long)NEDGES * HD;                   // 76.8M
  const int gEH = (int)((EH + 255) / 256);
  const int gNH = (NNODES * HD + 255) / 256;
  const int gRow8 = (NNODES + 7) / 8;
  const int gN256 = (NNODES + 255) / 256;

  // 1) h0 = relu(x @ fc_in_w + b)  -> h_path[:, 0:128]
  gemm_wmma_bf16<<<dim3(HD / BN, GM), 256, 0, stream>>>(
      x, CIN, fcw, HD, fcb, HP, LDIM, NNODES, CIN, ACT_RELU);

  // 2) three GENConv layers
  for (int i = 0; i < 3; i++) {
    const float* X = HP + (size_t)i * HD;  // conv input column block (stride LDIM)
    k_zero3<<<gNH, 256, 0, stream>>>(AGG, MXU, DEN, NNODES * HD);
    k_smax<<<gEH, 256, 0, stream>>>(X, LDIM, src, dst, MXU, ct + i, EH);
    k_ssum<<<gEH, 256, 0, stream>>>(X, LDIM, src, dst, MXU, DEN, AGG, ct + i, EH);
    k_finish_agg<<<gNH, 256, 0, stream>>>(AGG, DEN, X, LDIM, NNODES * HD);
    // MLP: Lin(128,256) -> LN -> ReLU -> Lin(256,128)
    gemm_wmma_bf16<<<dim3(2 * HD / BN, GM), 256, 0, stream>>>(
        AGG, HD, cw1 + (size_t)i * HD * 2 * HD, 2 * HD, cb1 + (size_t)i * 2 * HD,
        HMID, 2 * HD, NNODES, HD, ACT_NONE);
    k_ln_relu256<<<gRow8, 256, 0, stream>>>(HMID, cg1 + (size_t)i * 2 * HD,
                                            cbe1 + (size_t)i * 2 * HD, NNODES);
    float* outC = (i == 0) ? (HP + HD) : CONV;
    int ldcC = (i == 0) ? LDIM : HD;
    gemm_wmma_bf16<<<dim3(HD / BN, GM), 256, 0, stream>>>(
        HMID, 2 * HD, cw2 + (size_t)i * 2 * HD * HD, HD, cb2 + (size_t)i * HD,
        outC, ldcC, NNODES, 2 * HD, ACT_NONE);
    if (i >= 1) {  // h_{i+1} = relu(LN(c)) + h_i
      k_resid128<<<gRow8, 256, 0, stream>>>(CONV, dgg + (size_t)(i - 1) * HD,
                                            dgb + (size_t)(i - 1) * HD,
                                            HP + (size_t)i * HD,
                                            HP + (size_t)(i + 1) * HD, NNODES);
    }
  }

  // 3) attention head: AG = tanh(hp@wa+ba);  AG *= sigmoid(hp@wb+bb)
  gemm_wmma_bf16<<<dim3(LDIM / BN, GM), 256, 0, stream>>>(
      HP, LDIM, wa, LDIM, ba, AG, LDIM, NNODES, LDIM, ACT_TANH);
  gemm_wmma_bf16<<<dim3(LDIM / BN, GM), 256, 0, stream>>>(
      HP, LDIM, wb, LDIM, bb, AG, LDIM, NNODES, LDIM, ACT_SIGMUL);
  k_attnA<<<gRow8, 256, 0, stream>>>(AG, wc, bc, ARAW, NNODES);

  // 4) column softmax over N (per class) + pooled M = A @ h_path
  k_zero_stats<<<1, 256, 0, stream>>>(CMAX, CSUM, MB);
  k_cmax<<<gN256, 256, 0, stream>>>(ARAW, CMAX, NNODES);
  k_csum<<<gN256, 256, 0, stream>>>(ARAW, CMAX, CSUM, NNODES);
  k_wnorm<<<(NNODES * 4 + 255) / 256, 256, 0, stream>>>(ARAW, CMAX, CSUM, NNODES * 4);
  k_pool<<<gN256, 256, 0, stream>>>(HP, ARAW, MB, NNODES);

  // 5) type features + classifier + softmax/argmax
  k_tf<<<1, HD, 0, stream>>>(typ, tw, tb, TF);
  k_final<<<1, 128, 0, stream>>>(MB, TF, clw, clb, out);
}
